// GPT70B_32710470926764
// MI455X (gfx1250) — compile-verified, matrix operand order checked by reading, not run on hardware
//
#include <hip/hip_runtime.h>
#include <hip/hip_bf16.h>

typedef __bf16 v16bf __attribute__((ext_vector_type(16)));
typedef float  v8f   __attribute__((ext_vector_type(8)));

#define Tn 2048
#define En 1024
#define Hn 16
#define Ln 4
#define FFn 4096
#define Vn 32000
#define DHn 64

// ---------------- embedding: x = tok_emb[ids] + pos_emb ----------------
__global__ void embed_kernel(const int* __restrict__ ids,
                             const float* __restrict__ tok,
                             const float* __restrict__ pos,
                             float* __restrict__ x) {
  size_t i = (size_t)blockIdx.x * blockDim.x + threadIdx.x;
  if (i >= (size_t)Tn * En) return;
  int t = (int)(i / En), e = (int)(i % En);
  x[i] = tok[(size_t)ids[t] * En + e] + pos[i];
}

// ---------------- fp32 -> bf16 convert (weights) ----------------
__global__ void cvt_kernel(const float* __restrict__ s,
                           __hip_bfloat16* __restrict__ d, size_t n) {
  size_t i = (size_t)blockIdx.x * blockDim.x + threadIdx.x;
  if (i < n) d[i] = __float2bfloat16(s[i]);
}

// ---------------- layernorm (fp32 in, bf16 out) ----------------
__global__ void ln_kernel(const float* __restrict__ x,
                          const float* __restrict__ g,
                          const float* __restrict__ b,
                          __hip_bfloat16* __restrict__ out) {
  __shared__ float s1[256], s2[256];
  int t = blockIdx.x;
  const float* row = x + (size_t)t * En;
  float a = 0.f, q = 0.f;
  for (int e = threadIdx.x; e < En; e += 256) { float v = row[e]; a += v; q += v * v; }
  s1[threadIdx.x] = a; s2[threadIdx.x] = q; __syncthreads();
  for (int o = 128; o > 0; o >>= 1) {
    if (threadIdx.x < o) { s1[threadIdx.x] += s1[threadIdx.x + o]; s2[threadIdx.x] += s2[threadIdx.x + o]; }
    __syncthreads();
  }
  float m = s1[0] * (1.0f / En);
  float var = s2[0] * (1.0f / En) - m * m;
  float inv = rsqrtf(var + 1e-5f);
  for (int e = threadIdx.x; e < En; e += 256) {
    float v = (row[e] - m) * inv * g[e] + b[e];
    out[(size_t)t * En + e] = __float2bfloat16(v);
  }
}

// ---------------- V transpose: vT[h][d][t] = v[t][h*64+d] ----------------
__global__ void vtrans_kernel(const __hip_bfloat16* __restrict__ v,
                              __hip_bfloat16* __restrict__ vT) {
  size_t i = (size_t)blockIdx.x * blockDim.x + threadIdx.x;
  if (i >= (size_t)Tn * En) return;
  int t = (int)(i / En), c = (int)(i % En);
  int h = c >> 6, d = c & 63;
  vT[((size_t)h * DHn + d) * Tn + t] = v[i];
}

// ---------------- causal softmax row-wise, fp32 scores -> bf16 probs ----------------
__global__ void softmax_kernel(const float* __restrict__ sc,
                               __hip_bfloat16* __restrict__ pr) {
  __shared__ float red[256];
  int t = blockIdx.x;
  const float* row = sc + (size_t)t * Tn;
  float mx = -3.4e38f;
  for (int s = threadIdx.x; s <= t; s += 256) mx = fmaxf(mx, row[s]);
  red[threadIdx.x] = mx; __syncthreads();
  for (int o = 128; o > 0; o >>= 1) {
    if (threadIdx.x < o) red[threadIdx.x] = fmaxf(red[threadIdx.x], red[threadIdx.x + o]);
    __syncthreads();
  }
  mx = red[0]; __syncthreads();
  float sm = 0.f;
  for (int s = threadIdx.x; s <= t; s += 256) sm += __expf(row[s] - mx);
  red[threadIdx.x] = sm; __syncthreads();
  for (int o = 128; o > 0; o >>= 1) {
    if (threadIdx.x < o) red[threadIdx.x] += red[threadIdx.x + o];
    __syncthreads();
  }
  float inv = 1.0f / red[0];
  __hip_bfloat16* prow = pr + (size_t)t * Tn;
  for (int s = threadIdx.x; s < Tn; s += 256) {
    float v = (s <= t) ? __expf(row[s] - mx) * inv : 0.0f;
    prow[s] = __float2bfloat16(v);
  }
}

// ---------------- generic bf16 WMMA GEMM: C = act(scale*(A·B^T + bias)) + res ----
// A: M x K row-major (lda), B: N x K row-major (ldb). Block = 4 waves,
// wave tile 32x64, block tile 128x64. K multiple of 32; M,N multiples of tile.
// Register-double-buffered: step k+32's 6 fragments are loaded before the
// 8 WMMAs of step k execute, so wmma never waits on just-issued loads.
__global__ __launch_bounds__(128)
void gemm_bf16_kernel(const __hip_bfloat16* __restrict__ A, long lda,
                      const __hip_bfloat16* __restrict__ Bm, long ldb,
                      const float* __restrict__ bias, float scale, int gelu,
                      const float* __restrict__ res, long ldr,
                      float* __restrict__ outF, __hip_bfloat16* __restrict__ outB,
                      long ldc, int K, int causal) {
  int nBase = blockIdx.x * 64;
  int mBlk  = blockIdx.y * 128;
  if (causal && nBase > mBlk + 127) return;   // tile entirely above diagonal

  int wave = threadIdx.x >> 5;
  int lane = threadIdx.x & 31;
  int mBase = mBlk + wave * 32;
  int row = lane & 15;            // fragment row (A) / column (B)
  int kh  = (lane >> 4) << 4;     // K offset 0 or 16 within the 32-wide step

  v8f c[2][4];
#pragma unroll
  for (int s = 0; s < 2; ++s)
#pragma unroll
    for (int g = 0; g < 4; ++g)
      c[s][g] = (v8f){0.f, 0.f, 0.f, 0.f, 0.f, 0.f, 0.f, 0.f};

  const __hip_bfloat16* a0p = A + (size_t)(mBase + row) * lda + kh;
  const __hip_bfloat16* a1p = A + (size_t)(mBase + 16 + row) * lda + kh;
  const __hip_bfloat16* b0p = Bm + (size_t)(nBase + row) * ldb + kh;
  const __hip_bfloat16* b1p = b0p + (size_t)16 * ldb;
  const __hip_bfloat16* b2p = b0p + (size_t)32 * ldb;
  const __hip_bfloat16* b3p = b0p + (size_t)48 * ldb;

  // prologue loads (step k=0)
  v16bf a0 = *(const v16bf*)(a0p);
  v16bf a1 = *(const v16bf*)(a1p);
  v16bf b0 = *(const v16bf*)(b0p);
  v16bf b1 = *(const v16bf*)(b1p);
  v16bf b2 = *(const v16bf*)(b2p);
  v16bf b3 = *(const v16bf*)(b3p);

  for (int k = 32; k < K; k += 32) {
    // prefetch next-next cachelines while streaming (global_prefetch_b8)
    __builtin_prefetch(b0p + k + 128, 0, 3);
    __builtin_prefetch(b1p + k + 128, 0, 3);
    __builtin_prefetch(b2p + k + 128, 0, 3);
    __builtin_prefetch(b3p + k + 128, 0, 3);
    // issue step-k loads into the alternate register set
    v16bf na0 = *(const v16bf*)(a0p + k);
    v16bf na1 = *(const v16bf*)(a1p + k);
    v16bf nb0 = *(const v16bf*)(b0p + k);
    v16bf nb1 = *(const v16bf*)(b1p + k);
    v16bf nb2 = *(const v16bf*)(b2p + k);
    v16bf nb3 = *(const v16bf*)(b3p + k);
    // compute step k-32 on resident registers
    c[0][0] = __builtin_amdgcn_wmma_f32_16x16x32_bf16(false, a0, false, b0, (short)0, c[0][0], false, false);
    c[1][0] = __builtin_amdgcn_wmma_f32_16x16x32_bf16(false, a1, false, b0, (short)0, c[1][0], false, false);
    c[0][1] = __builtin_amdgcn_wmma_f32_16x16x32_bf16(false, a0, false, b1, (short)0, c[0][1], false, false);
    c[1][1] = __builtin_amdgcn_wmma_f32_16x16x32_bf16(false, a1, false, b1, (short)0, c[1][1], false, false);
    c[0][2] = __builtin_amdgcn_wmma_f32_16x16x32_bf16(false, a0, false, b2, (short)0, c[0][2], false, false);
    c[1][2] = __builtin_amdgcn_wmma_f32_16x16x32_bf16(false, a1, false, b2, (short)0, c[1][2], false, false);
    c[0][3] = __builtin_amdgcn_wmma_f32_16x16x32_bf16(false, a0, false, b3, (short)0, c[0][3], false, false);
    c[1][3] = __builtin_amdgcn_wmma_f32_16x16x32_bf16(false, a1, false, b3, (short)0, c[1][3], false, false);
    a0 = na0; a1 = na1; b0 = nb0; b1 = nb1; b2 = nb2; b3 = nb3;
  }
  // epilogue compute (last k step)
  c[0][0] = __builtin_amdgcn_wmma_f32_16x16x32_bf16(false, a0, false, b0, (short)0, c[0][0], false, false);
  c[1][0] = __builtin_amdgcn_wmma_f32_16x16x32_bf16(false, a1, false, b0, (short)0, c[1][0], false, false);
  c[0][1] = __builtin_amdgcn_wmma_f32_16x16x32_bf16(false, a0, false, b1, (short)0, c[0][1], false, false);
  c[1][1] = __builtin_amdgcn_wmma_f32_16x16x32_bf16(false, a1, false, b1, (short)0, c[1][1], false, false);
  c[0][2] = __builtin_amdgcn_wmma_f32_16x16x32_bf16(false, a0, false, b2, (short)0, c[0][2], false, false);
  c[1][2] = __builtin_amdgcn_wmma_f32_16x16x32_bf16(false, a1, false, b2, (short)0, c[1][2], false, false);
  c[0][3] = __builtin_amdgcn_wmma_f32_16x16x32_bf16(false, a0, false, b3, (short)0, c[0][3], false, false);
  c[1][3] = __builtin_amdgcn_wmma_f32_16x16x32_bf16(false, a1, false, b3, (short)0, c[1][3], false, false);

  // D layout: VGPR r -> M = r + 8*(lane>=16), N = lane&15
  int mOff = (lane >> 4) * 8;
  int nOff = lane & 15;
#pragma unroll
  for (int s = 0; s < 2; ++s)
#pragma unroll
    for (int g = 0; g < 4; ++g)
#pragma unroll
      for (int r = 0; r < 8; ++r) {
        int m = mBase + s * 16 + mOff + r;
        int n = nBase + g * 16 + nOff;
        float v = c[s][g][r];
        if (bias) v += bias[n];
        v *= scale;
        if (gelu) v = 0.5f * v * (1.0f + erff(v * 0.70710678118f));
        if (res)  v += res[(size_t)m * ldr + n];
        if (outF) outF[(size_t)m * ldc + n] = v;
        if (outB) outB[(size_t)m * ldc + n] = __float2bfloat16(v);
      }
}

// ======================= host side =======================
extern "C" void kernel_launch(void* const* d_in, const int* in_sizes, int n_in,
                              void* d_out, int out_size, void* d_ws, size_t ws_size,
                              hipStream_t stream) {
  const int*   ids  = (const int*)d_in[0];
  const float* tok  = (const float*)d_in[1];
  const float* pos  = (const float*)d_in[2];
  const float* Wq   = (const float*)d_in[3];  const float* bq = (const float*)d_in[4];
  const float* Wk   = (const float*)d_in[5];  const float* bk = (const float*)d_in[6];
  const float* Wv   = (const float*)d_in[7];  const float* bv = (const float*)d_in[8];
  const float* Wo   = (const float*)d_in[9];  const float* bo = (const float*)d_in[10];
  const float* g1   = (const float*)d_in[11]; const float* be1 = (const float*)d_in[12];
  const float* W1   = (const float*)d_in[13]; const float* b1 = (const float*)d_in[14];
  const float* W2   = (const float*)d_in[15]; const float* b2 = (const float*)d_in[16];
  const float* g2   = (const float*)d_in[17]; const float* be2 = (const float*)d_in[18];
  const float* gF   = (const float*)d_in[19]; const float* beF = (const float*)d_in[20];
  const float* Wh   = (const float*)d_in[21];
  float* logits = (float*)d_out;

  // workspace layout (bytes)
  size_t off = 0;
  auto alloc = [&](size_t bytes) { size_t o = off; off += (bytes + 255) & ~(size_t)255; return o; };
  size_t oX  = alloc((size_t)Tn * En * 4);   // residual stream fp32
  size_t oH  = alloc((size_t)Tn * En * 2);   // LN output bf16
  size_t oQ  = alloc((size_t)Tn * En * 2);
  size_t oK  = alloc((size_t)Tn * En * 2);
  size_t oV  = alloc((size_t)Tn * En * 2);
  size_t oVT = alloc((size_t)Tn * En * 2);
  size_t oO  = alloc((size_t)Tn * En * 2);   // attn output bf16
  size_t oF  = alloc((size_t)Tn * FFn * 2);  // FF hidden bf16
  size_t oS  = alloc((size_t)Tn * Tn * 4);   // scores fp32 (one head)
  size_t oP  = alloc((size_t)Tn * Tn * 2);   // probs bf16 (one head)
  size_t oW  = alloc((size_t)8192000 * 2);   // bf16 weight staging (max 8000x1024)
  if (ws_size < off) return;

  char* ws = (char*)d_ws;
  float* x            = (float*)(ws + oX);
  __hip_bfloat16* h   = (__hip_bfloat16*)(ws + oH);
  __hip_bfloat16* q   = (__hip_bfloat16*)(ws + oQ);
  __hip_bfloat16* k   = (__hip_bfloat16*)(ws + oK);
  __hip_bfloat16* v   = (__hip_bfloat16*)(ws + oV);
  __hip_bfloat16* vT  = (__hip_bfloat16*)(ws + oVT);
  __hip_bfloat16* ob  = (__hip_bfloat16*)(ws + oO);
  __hip_bfloat16* f   = (__hip_bfloat16*)(ws + oF);
  float* sc           = (float*)(ws + oS);
  __hip_bfloat16* pr  = (__hip_bfloat16*)(ws + oP);
  __hip_bfloat16* wb  = (__hip_bfloat16*)(ws + oW);

  auto cvt = [&](const float* s, __hip_bfloat16* d, size_t n) {
    cvt_kernel<<<(unsigned)((n + 255) / 256), 256, 0, stream>>>(s, d, n);
  };
  auto gemm = [&](const __hip_bfloat16* A, long lda, const __hip_bfloat16* Bm, long ldb,
                  const float* bias, float scale, int gelu,
                  const float* res, long ldr,
                  float* outFp, __hip_bfloat16* outBp, long ldc,
                  int M, int N, int Kd, int causal) {
    dim3 grid((unsigned)(N / 64), (unsigned)(M / 128));
    gemm_bf16_kernel<<<grid, 128, 0, stream>>>(A, lda, Bm, ldb, bias, scale, gelu,
                                               res, ldr, outFp, outBp, ldc, Kd, causal);
  };

  // ---- embedding ----
  embed_kernel<<<(Tn * En + 255) / 256, 256, 0, stream>>>(ids, tok, pos, x);

  const float qscale = 0.125f; // 1/sqrt(DH)

  for (int l = 0; l < Ln; ++l) {
    const size_t wEE = (size_t)En * En;
    // LN1
    ln_kernel<<<Tn, 256, 0, stream>>>(x, g1 + (size_t)l * En, be1 + (size_t)l * En, h);
    // Q/K/V projections (weights converted to bf16 on the fly)
    cvt(Wq + (size_t)l * wEE, wb, wEE);
    gemm(h, En, wb, En, bq + (size_t)l * En, qscale, 0, nullptr, 0, nullptr, q, En, Tn, En, En, 0);
    cvt(Wk + (size_t)l * wEE, wb, wEE);
    gemm(h, En, wb, En, bk + (size_t)l * En, 1.0f, 0, nullptr, 0, nullptr, k, En, Tn, En, En, 0);
    cvt(Wv + (size_t)l * wEE, wb, wEE);
    gemm(h, En, wb, En, bv + (size_t)l * En, 1.0f, 0, nullptr, 0, nullptr, v, En, Tn, En, En, 0);
    vtrans_kernel<<<(Tn * En + 255) / 256, 256, 0, stream>>>(v, vT);

    // attention, one head at a time (bounded scratch)
    for (int hd = 0; hd < Hn; ++hd) {
      // scores = (q/8) · k^T  (causal blocks skipped)
      gemm(q + hd * DHn, En, k + hd * DHn, En, nullptr, 1.0f, 0, nullptr, 0,
           sc, nullptr, Tn, Tn, Tn, DHn, 1);
      softmax_kernel<<<Tn, 256, 0, stream>>>(sc, pr);
      // o_head = probs · v_head  (vT is DH x T row-major = N x K)
      gemm(pr, Tn, vT + (size_t)hd * DHn * Tn, Tn, nullptr, 1.0f, 0, nullptr, 0,
           nullptr, ob + hd * DHn, En, Tn, DHn, Tn, 0);
    }
    // output projection + residual into x
    cvt(Wo + (size_t)l * wEE, wb, wEE);
    gemm(ob, En, wb, En, bo + (size_t)l * En, 1.0f, 0, x, En, x, nullptr, En, Tn, En, En, 0);

    // MLP
    ln_kernel<<<Tn, 256, 0, stream>>>(x, g2 + (size_t)l * En, be2 + (size_t)l * En, h);
    cvt(W1 + (size_t)l * FFn * En, wb, (size_t)FFn * En);
    gemm(h, En, wb, En, b1 + (size_t)l * FFn, 1.0f, 1, nullptr, 0, nullptr, f, FFn, Tn, FFn, En, 0);
    cvt(W2 + (size_t)l * En * FFn, wb, (size_t)En * FFn);
    gemm(f, FFn, wb, FFn, b2 + (size_t)l * En, 1.0f, 0, x, En, x, nullptr, En, Tn, En, FFn, 0);
  }

  // final LN + LM head (chunked over vocab)
  ln_kernel<<<Tn, 256, 0, stream>>>(x, gF, beF, h);
  const int CH = 8000; // 4 chunks of 8000 (125 x 64-wide tiles each)
  for (int c0 = 0; c0 < Vn; c0 += CH) {
    cvt(Wh + (size_t)c0 * En, wb, (size_t)CH * En);
    gemm(h, En, wb, En, nullptr, 1.0f, 0, nullptr, 0,
         logits + c0, nullptr, Vn, Tn, CH, En, 0);
  }
}
